// ssse_21672404975787
// MI455X (gfx1250) — matrix operand order repaired and checked by reading, not compiled, not run on hardware
//
#include <hip/hip_runtime.h>

#define HW 65536
#define EPS 1e-5f

typedef __attribute__((ext_vector_type(16))) __bf16 v16bf;
typedef __attribute__((ext_vector_type(2)))  __bf16 v2bf;
typedef __attribute__((ext_vector_type(8)))  float  v8f;

// native f32->bf16 (RNE) conversions; backend picks v_cvt_*_bf16_f32
__device__ __forceinline__ unsigned short f2bf(float f) {
  union { __bf16 h; unsigned short u; } x;
  x.h = (__bf16)f;
  return x.u;
}
__device__ __forceinline__ unsigned int pk2(float a, float b) {
  union { v2bf v; unsigned int u; } x;
  x.v[0] = (__bf16)a;
  x.v[1] = (__bf16)b;
  return x.u;
}
// dword holds bf16 pair (lo=ch0, hi=ch1); apply per-channel affine + relu, repack
__device__ __forceinline__ unsigned int bn2(unsigned int u, float s0, float h0,
                                            float s1, float h1) {
  union { unsigned int u; float f; } lo, hi;
  lo.u = u << 16;
  hi.u = u & 0xFFFF0000u;
  float a = fmaf(lo.f, s0, h0); a = a > 0.f ? a : 0.f;
  float b = fmaf(hi.f, s1, h1); b = b > 0.f ? b : 0.f;
  return pk2(a, b);
}

// ---------------------------------------------------------------- utilities
__global__ void k_zero(float* p, int n) {
  int t = blockIdx.x * 256 + threadIdx.x;
  if (t < n) p[t] = 0.f;
}

// Finalize BN affine: scale = g*rsqrt(var+eps), shift = b - mean*scale
__global__ void k_stats(const float* __restrict__ sum, const float* __restrict__ sq,
                        const float* __restrict__ gam, const float* __restrict__ bet,
                        float* __restrict__ scale, float* __restrict__ shift,
                        int C, float invN) {
  int c = threadIdx.x;
  if (c < C) {
    float m = sum[c] * invN;
    float v = sq[c] * invN - m * m;
    float sc = gam[c] * rsqrtf(v + EPS);
    scale[c] = sc;
    shift[c] = bet[c] - m * sc;
  }
}

// Per-channel sum / sumsq over a [b][p][C] bf16 buffer (C = 32 or 64)
template <int C>
__global__ void k_redstats(const unsigned short* __restrict__ y,
                           float* __restrict__ sum, float* __restrict__ sq) {
  const int G = C / 8;  // dword-quad groups per pixel
  __shared__ float s_sum[C], s_sq[C];
  int tid = threadIdx.x;
  if (tid < C) { s_sum[tid] = 0.f; s_sq[tid] = 0.f; }
  __syncthreads();
  float a[8], b[8];
#pragma unroll
  for (int i = 0; i < 8; i++) { a[i] = 0.f; b[i] = 0.f; }
  const long total = 2097152L * G;  // B*HW units of 8 channels
  for (long u = (long)blockIdx.x * 256 + tid; u < total; u += (long)gridDim.x * 256) {
    uint4 q = *(const uint4*)(y + u * 8);
    unsigned int d[4] = {q.x, q.y, q.z, q.w};
#pragma unroll
    for (int i = 0; i < 4; i++) {
      float lo = __uint_as_float(d[i] << 16);
      float hi = __uint_as_float(d[i] & 0xFFFF0000u);
      a[2 * i] += lo;     b[2 * i] += lo * lo;
      a[2 * i + 1] += hi; b[2 * i + 1] += hi * hi;
    }
  }
  int cg = tid & (G - 1);  // constant across grid-stride since 256 % G == 0
#pragma unroll
  for (int i = 0; i < 8; i++)
    for (int off = G; off < 32; off <<= 1) {
      a[i] += __shfl_xor(a[i], off);
      b[i] += __shfl_xor(b[i], off);
    }
  if ((tid & 31) < G) {
#pragma unroll
    for (int i = 0; i < 8; i++) {
      atomicAdd(&s_sum[cg * 8 + i], a[i]);
      atomicAdd(&s_sq[cg * 8 + i], b[i]);
    }
  }
  __syncthreads();
  if (tid < C) { atomicAdd(&sum[tid], s_sum[tid]); atomicAdd(&sq[tid], s_sq[tid]); }
}

// ----------------------------------------------------- weight repack (bf16)
// A-fragment layout (16x32 bf16, wave32): lane l holds row M=l&15.
// vector elem j -> K = j + (l>=16 ? 16:0)
__global__ void k_repack(const float* __restrict__ w2,   // (64,32)
                         const float* __restrict__ wsa1, // (32,64,3,3)
                         const float* __restrict__ wsa2, // (32,32,3,3)
                         const float* __restrict__ wf1,  // (32,32)
                         const float* __restrict__ wf2,  // (16,32)
                         unsigned short* __restrict__ wpk) {
  const int C3 = 2048, C4 = 20480, F1 = 29696, F2 = 30720, TOT = 31232;
  for (int idx = blockIdx.x * 256 + threadIdx.x; idx < TOT; idx += gridDim.x * 256) {
    int within = idx & 511;
    int l = within >> 4;
    int j = within & 15;
    int v = j >> 1, h = j & 1;
    int K = (2 * v + h) + ((l >= 16) ? 16 : 0);
    int mloc = l & 15;
    float val;
    if (idx < C3) {                       // conv2
      int mt = idx >> 9;
      val = w2[(mt * 16 + mloc) * 32 + K];
    } else if (idx < C4) {                // conv3, Cin=64
      int f = (idx - C3) >> 9;
      int ks = f >> 1, mt = f & 1;
      int k = ks * 32 + K;
      int c = k & 63, s = k >> 6;
      int ky = s / 3, kx = s % 3;
      val = wsa1[(((mt * 16 + mloc) * 64 + c) * 3 + ky) * 3 + kx];
    } else if (idx < F1) {                // conv4, Cin=32
      int f = (idx - C4) >> 9;
      int ks = f >> 1, mt = f & 1;
      int k = ks * 32 + K;
      int c = k & 31, s = k >> 5;
      int ky = s / 3, kx = s % 3;
      val = wsa2[(((mt * 16 + mloc) * 32 + c) * 3 + ky) * 3 + kx];
    } else if (idx < F2) {                // fc1
      int mt = (idx - F1) >> 9;
      val = wf1[(mt * 16 + mloc) * 32 + K];
    } else {                              // fc2 (M=16)
      val = wf2[mloc * 32 + K];
    }
    wpk[idx] = f2bf(val);
  }
}

// -------------------------- conv1: 1x1 16->32 scalar, out [b][p][32] bf16
__global__ void k_conv1(const float* __restrict__ x, const float* __restrict__ w,
                        unsigned short* __restrict__ y1) {
  __shared__ float sw[512];
  int tid = threadIdx.x;
  for (int i = tid; i < 512; i += 256) sw[i] = w[i];
  __syncthreads();
  long g = (long)blockIdx.x * 256 + tid;
  int b = (int)(g >> 16);
  int p = (int)(g & 65535);
  const float* xb = x + ((long)b * 16) * HW + p;
  float in[16];
#pragma unroll
  for (int c = 0; c < 16; c++) in[c] = xb[(long)c * HW];
  float acc[32];
#pragma unroll
  for (int m = 0; m < 32; m++) {
    float a = 0.f;
#pragma unroll
    for (int c = 0; c < 16; c++) a = fmaf(sw[m * 16 + c], in[c], a);
    acc[m] = a;
  }
  unsigned int q[16];
#pragma unroll
  for (int d = 0; d < 16; d++) q[d] = pk2(acc[2 * d], acc[2 * d + 1]);
  uint4* dst = (uint4*)(y1 + ((long)b * HW + p) * 32);
  dst[0] = *(uint4*)&q[0];
  dst[1] = *(uint4*)&q[4];
  dst[2] = *(uint4*)&q[8];
  dst[3] = *(uint4*)&q[12];
}

// -------------------- conv2: 1x1 32->64 (WMMA bf16), [p][32] -> [p][64]
__global__ void k_conv2(const unsigned short* __restrict__ yin,
                        const float* __restrict__ scale, const float* __restrict__ shift,
                        const unsigned short* __restrict__ wpk,
                        unsigned short* __restrict__ yout) {
  const int STR = 48;
  __shared__ __align__(32) unsigned short lin[256 * STR];
  __shared__ float s_scale[32], s_shift[32];
  int tid = threadIdx.x;
  if (tid < 32) { s_scale[tid] = scale[tid]; s_shift[tid] = shift[tid]; }
  __syncthreads();
  long g0 = (long)blockIdx.x * 256;
  int b = (int)(g0 >> 16);
  int p0 = (int)(g0 & 65535);
  {
    const uint4* src = (const uint4*)(yin + ((long)b * HW + p0 + tid) * 32);
    unsigned int q[16];
    *(uint4*)&q[0] = src[0];
    *(uint4*)&q[4] = src[1];
    *(uint4*)&q[8] = src[2];
    *(uint4*)&q[12] = src[3];
#pragma unroll
    for (int d = 0; d < 16; d++)
      q[d] = bn2(q[d], s_scale[2 * d], s_shift[2 * d], s_scale[2 * d + 1], s_shift[2 * d + 1]);
    *(uint4*)&lin[tid * STR + 0]  = *(uint4*)&q[0];
    *(uint4*)&lin[tid * STR + 8]  = *(uint4*)&q[4];
    *(uint4*)&lin[tid * STR + 16] = *(uint4*)&q[8];
    *(uint4*)&lin[tid * STR + 24] = *(uint4*)&q[12];
  }
  __syncthreads();
  int w = tid >> 5, lane = tid & 31, l16 = lane & 15, hi = lane >> 4;
  for (int nt = 2 * w; nt <= 2 * w + 1; ++nt) {
    int pxb = nt * 16;
    v16bf bfrag = *(const v16bf*)(const void*)(&lin[(pxb + l16) * STR + hi * 16]);
    long pbase = ((long)b * HW + p0 + pxb + l16) * 64;
    for (int mt = 0; mt < 4; ++mt) {
      v16bf afrag = *(const v16bf*)(const void*)(&wpk[mt * 512 + lane * 16]);
      v8f acc = {};
      acc = __builtin_amdgcn_wmma_f32_16x16x32_bf16(false, afrag, false, bfrag,
                                                    (short)0, acc, false, false);
      unsigned int q[4];
#pragma unroll
      for (int r = 0; r < 4; ++r) q[r] = pk2(acc[2 * r], acc[2 * r + 1]);
      *(uint4*)(yout + pbase + mt * 16 + hi * 8) = *(uint4*)&q[0];
    }
  }
}

// --------------------------- 3x3 conv, implicit GEMM, Cout=32 (WMMA bf16)
template <int CIN, int KSTEPS>
__global__ void k_conv3x3(const unsigned short* __restrict__ yin,
                          const float* __restrict__ scale, const float* __restrict__ shift,
                          const unsigned short* __restrict__ wpk,
                          unsigned short* __restrict__ yout) {
  const int STR = CIN + 16;  // pixel stride (elems): keeps 16B/32B alignment, de-banks
  const int G = CIN / 8;
  __shared__ __align__(32) unsigned short lin[6 * 66 * STR];
  __shared__ float s_scale[CIN], s_shift[CIN];
  int tid = threadIdx.x;
  if (tid < CIN) { s_scale[tid] = scale[tid]; s_shift[tid] = shift[tid]; }
  __syncthreads();
  int bi = blockIdx.x;
  int b = bi >> 8;
  int rem = bi & 255;
  int row0 = (rem >> 2) * 4;
  int col0 = (rem & 3) * 64;
  for (int unit = tid; unit < 6 * 66 * G; unit += 256) {
    int cg = unit & (G - 1);
    int pix = unit / G;
    int ci = pix % 66;
    int ri = pix / 66;
    int y = row0 - 1 + ri;
    int xx = col0 - 1 + ci;
    unsigned int q[4] = {0u, 0u, 0u, 0u};
    if ((unsigned)y < 256u && (unsigned)xx < 256u) {
      *(uint4*)q = *(const uint4*)(yin + ((long)b * HW + y * 256 + xx) * CIN + cg * 8);
      int c0 = cg * 8;
#pragma unroll
      for (int d = 0; d < 4; ++d)
        q[d] = bn2(q[d], s_scale[c0 + 2 * d], s_shift[c0 + 2 * d],
                   s_scale[c0 + 2 * d + 1], s_shift[c0 + 2 * d + 1]);
    }
    *(uint4*)&lin[(ri * 66 + ci) * STR + cg * 8] = *(uint4*)q;
  }
  __syncthreads();
  int w = tid >> 5, lane = tid & 31, l16 = lane & 15, hi = lane >> 4;
  int s0 = 2 * w, s1 = 2 * w + 1;
  int r_a = s0 >> 2, cg_a = s0 & 3;
  int r_b = s1 >> 2, cg_b = s1 & 3;
  v8f acc[2][2] = {};
  for (int ks = 0; ks < KSTEPS; ++ks) {
    int sidx = (CIN == 64) ? (ks >> 1) : ks;
    int cbase = (CIN == 64) ? ((ks & 1) * 32) : 0;
    int ky = sidx / 3, kx = sidx % 3;
    v16bf a0 = *(const v16bf*)(const void*)(&wpk[(ks * 2 + 0) * 512 + lane * 16]);
    v16bf a1 = *(const v16bf*)(const void*)(&wpk[(ks * 2 + 1) * 512 + lane * 16]);
    int co = cbase + hi * 16;
    v16bf b0 = *(const v16bf*)(const void*)(&lin[((r_a + ky) * 66 + cg_a * 16 + l16 + kx) * STR + co]);
    v16bf b1 = *(const v16bf*)(const void*)(&lin[((r_b + ky) * 66 + cg_b * 16 + l16 + kx) * STR + co]);
    acc[0][0] = __builtin_amdgcn_wmma_f32_16x16x32_bf16(false, a0, false, b0, (short)0, acc[0][0], false, false);
    acc[0][1] = __builtin_amdgcn_wmma_f32_16x16x32_bf16(false, a1, false, b0, (short)0, acc[0][1], false, false);
    acc[1][0] = __builtin_amdgcn_wmma_f32_16x16x32_bf16(false, a0, false, b1, (short)0, acc[1][0], false, false);
    acc[1][1] = __builtin_amdgcn_wmma_f32_16x16x32_bf16(false, a1, false, b1, (short)0, acc[1][1], false, false);
  }
#pragma unroll
  for (int ntl = 0; ntl < 2; ++ntl) {
    int rr = ntl ? r_b : r_a;
    int cg = ntl ? cg_b : cg_a;
    int p = (row0 + rr) * 256 + col0 + cg * 16 + l16;
    long pbase = ((long)b * HW + p) * 32;
#pragma unroll
    for (int mt = 0; mt < 2; ++mt) {
      unsigned int q[4];
#pragma unroll
      for (int r = 0; r < 4; ++r) q[r] = pk2(acc[ntl][mt][2 * r], acc[ntl][mt][2 * r + 1]);
      *(uint4*)(yout + pbase + mt * 16 + hi * 8) = *(uint4*)&q[0];
    }
  }
}

// ----------------------- per-pixel MLP: tanh(W2 tanh(W1 v + b1) + b2), sum
__global__ void k_fc(const unsigned short* __restrict__ y4,
                     const float* __restrict__ scale, const float* __restrict__ shift,
                     const unsigned short* __restrict__ wpk_f1,
                     const unsigned short* __restrict__ wpk_f2,
                     const float* __restrict__ bf1, const float* __restrict__ bf2,
                     float* __restrict__ y0sum) {
  const int STR = 48;
  __shared__ __align__(32) unsigned short vtile[256 * STR];
  __shared__ __align__(32) unsigned short atile[256 * STR];
  __shared__ float s_scale[32], s_shift[32], s_b1[32], s_b2[16], s_acc[16];
  int tid = threadIdx.x;
  if (tid < 32) { s_scale[tid] = scale[tid]; s_shift[tid] = shift[tid]; s_b1[tid] = bf1[tid]; }
  if (tid < 16) { s_b2[tid] = bf2[tid]; s_acc[tid] = 0.f; }
  __syncthreads();
  long g0 = (long)blockIdx.x * 256;
  int b = (int)(g0 >> 16);
  int p0 = (int)(g0 & 65535);
  {
    const uint4* src = (const uint4*)(y4 + ((long)b * HW + p0 + tid) * 32);
    unsigned int q[16];
    *(uint4*)&q[0] = src[0];
    *(uint4*)&q[4] = src[1];
    *(uint4*)&q[8] = src[2];
    *(uint4*)&q[12] = src[3];
#pragma unroll
    for (int d = 0; d < 16; d++)
      q[d] = bn2(q[d], s_scale[2 * d], s_shift[2 * d], s_scale[2 * d + 1], s_shift[2 * d + 1]);
    *(uint4*)&vtile[tid * STR + 0]  = *(uint4*)&q[0];
    *(uint4*)&vtile[tid * STR + 8]  = *(uint4*)&q[4];
    *(uint4*)&vtile[tid * STR + 16] = *(uint4*)&q[8];
    *(uint4*)&vtile[tid * STR + 24] = *(uint4*)&q[12];
  }
  __syncthreads();
  int w = tid >> 5, lane = tid & 31, l16 = lane & 15, hi = lane >> 4;
  for (int nt = 2 * w; nt <= 2 * w + 1; ++nt) {
    int pxb = nt * 16;
    v16bf bfrag = *(const v16bf*)(const void*)(&vtile[(pxb + l16) * STR + hi * 16]);
#pragma unroll
    for (int mt = 0; mt < 2; ++mt) {
      v16bf afrag = *(const v16bf*)(const void*)(&wpk_f1[mt * 512 + lane * 16]);
      v8f cfrag;
#pragma unroll
      for (int r = 0; r < 8; ++r) cfrag[r] = s_b1[mt * 16 + r + hi * 8];
      v8f acc = __builtin_amdgcn_wmma_f32_16x16x32_bf16(false, afrag, false, bfrag,
                                                        (short)0, cfrag, false, false);
#pragma unroll
      for (int r = 0; r < 4; ++r) {
        unsigned int qq = pk2(tanhf(acc[2 * r]), tanhf(acc[2 * r + 1]));
        *(unsigned int*)&atile[(pxb + l16) * STR + mt * 16 + 2 * r + hi * 8] = qq;
      }
    }
  }
  __syncthreads();
  float uacc[8];
#pragma unroll
  for (int r = 0; r < 8; ++r) uacc[r] = 0.f;
  for (int nt = 2 * w; nt <= 2 * w + 1; ++nt) {
    int pxb = nt * 16;
    v16bf bfrag = *(const v16bf*)(const void*)(&atile[(pxb + l16) * STR + hi * 16]);
    v16bf afrag = *(const v16bf*)(const void*)(&wpk_f2[lane * 16]);
    v8f cfrag;
#pragma unroll
    for (int r = 0; r < 8; ++r) cfrag[r] = s_b2[r + hi * 8];
    v8f acc = __builtin_amdgcn_wmma_f32_16x16x32_bf16(false, afrag, false, bfrag,
                                                      (short)0, cfrag, false, false);
#pragma unroll
    for (int r = 0; r < 8; ++r) uacc[r] += tanhf(acc[r]);
  }
#pragma unroll
  for (int r = 0; r < 8; ++r) atomicAdd(&s_acc[r + hi * 8], uacc[r]);
  __syncthreads();
  if (tid < 16) atomicAdd(&y0sum[b * 16 + tid], s_acc[tid]);
}

// ---------------------------------------- per-batch head: Cm, norms, sort
__global__ void k_head(const float* __restrict__ y0sum, float* __restrict__ cm_ws,
                       int* __restrict__ gi_ws, float* __restrict__ out_wi,
                       float* __restrict__ out_vals, float* __restrict__ out_order) {
  __shared__ float y0[16], cm[256], wn[16], vals[16];
  __shared__ int ord[16];
  int b = blockIdx.x;
  int t = threadIdx.x;  // 32 threads
  if (t < 16) y0[t] = y0sum[b * 16 + t] * (1.0f / 65536.0f);
  __syncthreads();
  for (int e = t; e < 256; e += 32) {
    int i = e >> 4, j = e & 15;
    float v = (i == j) ? 0.f : y0[i] * y0[j];
    cm[e] = v;
    cm_ws[b * 256 + e] = v;
  }
  __syncthreads();
  if (t < 16) {
    float s = 0.f;
    for (int j = 0; j < 16; ++j) s += cm[t * 16 + j] * cm[t * 16 + j];
    wn[t] = sqrtf(s);
  }
  __syncthreads();
  if (t == 0) {  // descending selection sort (matches argsort(-w))
    float v[16]; int id[16];
    for (int i = 0; i < 16; i++) { v[i] = wn[i]; id[i] = i; }
    for (int i = 0; i < 16; i++) {
      int best = i;
      for (int j = i + 1; j < 16; j++) if (v[j] > v[best]) best = j;
      float tv = v[i]; v[i] = v[best]; v[best] = tv;
      int ti = id[i]; id[i] = id[best]; id[best] = ti;
      vals[i] = v[i]; ord[i] = id[i];
    }
  }
  __syncthreads();
  if (t < 16) {
    out_vals[b * 16 + t] = vals[t];
    out_order[b * 16 + t] = (float)ord[t];
    gi_ws[b * 16 + t] = ord[t];
  }
  if (t < 5)
    out_wi[t * 32 + b] = (vals[3 * t] + vals[3 * t + 1] + vals[3 * t + 2]) * (1.f / 3.f);
}

// --------------------------------------------------- gather fi = x[b, gi]
__global__ void k_sel(const float* __restrict__ x, const int* __restrict__ gi,
                      float* __restrict__ out_fi) {
  long t = (long)blockIdx.x * 256 + threadIdx.x;  // total = 32*15*HW
  int p = (int)(t & 65535);
  long r = t >> 16;
  int j15 = (int)(r % 15);
  int b = (int)(r / 15);
  int ch = gi[b * 16 + j15];
  int i = j15 / 3, jj = j15 % 3;
  long off = (long)i * (32L * 3 * HW) + ((long)(b * 3 + jj)) * HW + p;
  out_fi[off] = x[((long)(b * 16 + ch)) * HW + p];
}

// -------------------------------------------- cgh = Cm (16x16) @ x (16xHW)
__global__ void k_cgh(const float* __restrict__ x, const float* __restrict__ cm_ws,
                      float* __restrict__ out_cgh) {
  __shared__ float scm[256];
  int tid = threadIdx.x;
  long g = (long)blockIdx.x * 256 + tid;
  int b = (int)(g >> 16);
  int p = (int)(g & 65535);
  scm[tid] = cm_ws[b * 256 + tid];
  __syncthreads();
  const float* xb = x + ((long)b * 16) * HW + p;
  float xv[16];
#pragma unroll
  for (int j = 0; j < 16; j++) xv[j] = xb[(long)j * HW];
  float* ob = out_cgh + ((long)b * 16) * HW + p;
#pragma unroll
  for (int i = 0; i < 16; i++) {
    float a = 0.f;
#pragma unroll
    for (int j = 0; j < 16; j++) a = fmaf(scm[i * 16 + j], xv[j], a);
    ob[(long)i * HW] = a;
  }
}

extern "C" void kernel_launch(void* const* d_in, const int* in_sizes, int n_in,
                              void* d_out, int out_size, void* d_ws, size_t ws_size,
                              hipStream_t stream) {
  (void)in_sizes; (void)n_in; (void)out_size; (void)ws_size;
  const float* x    = (const float*)d_in[0];
  const float* w1   = (const float*)d_in[1];
  const float* g1   = (const float*)d_in[2];
  const float* bb1  = (const float*)d_in[3];
  const float* w2   = (const float*)d_in[4];
  const float* g2   = (const float*)d_in[5];
  const float* bb2  = (const float*)d_in[6];
  const float* wsa1 = (const float*)d_in[7];
  const float* g3   = (const float*)d_in[8];
  const float* bb3  = (const float*)d_in[9];
  const float* wsa2 = (const float*)d_in[10];
  const float* g4   = (const float*)d_in[11];
  const float* bb4  = (const float*)d_in[12];
  const float* wf1  = (const float*)d_in[13];
  const float* bf1  = (const float*)d_in[14];
  const float* wf2  = (const float*)d_in[15];
  const float* bf2  = (const float*)d_in[16];
  float* out = (float*)d_out;

  unsigned char* ws = (unsigned char*)d_ws;
  unsigned short* yA = (unsigned short*)ws;                               // 256 MiB ([p][64]: y2 then y4)
  unsigned short* yB = (unsigned short*)(ws + (size_t)268435456);         // 128 MiB ([p][32]: y1 then y3)
  float* stats = (float*)(ws + (size_t)268435456 + (size_t)134217728);    // 4 stages * 256 f32
  float* y0sum = stats + 1024;                                            // 512 f32
  float* cmws  = y0sum + 512;                                             // 8192 f32
  int*   giws  = (int*)(cmws + 8192);                                     // 512 i32
  unsigned short* wpk = (unsigned short*)(giws + 512);                    // 31232 bf16

  const size_t OFF_WI = 31457280, OFF_VALS = OFF_WI + 160,
               OFF_ORDER = OFF_VALS + 512, OFF_CGH = OFF_ORDER + 512;
  const float invN = 1.0f / 2097152.0f;  // B*H*W

  float* st0 = stats;           // conv1 (sum, sq, scale, shift @ +0,+64,+128,+192)
  float* st1 = stats + 256;     // conv2
  float* st2 = stats + 512;     // conv3
  float* st3 = stats + 768;     // conv4

  k_zero<<<6, 256, 0, stream>>>(stats, 1536);  // stats sums + y0sum
  k_repack<<<64, 256, 0, stream>>>(w2, wsa1, wsa2, wf1, wf2, wpk);

  k_conv1<<<8192, 256, 0, stream>>>(x, w1, yB);
  k_redstats<32><<<2048, 256, 0, stream>>>(yB, st0, st0 + 64);
  k_stats<<<1, 64, 0, stream>>>(st0, st0 + 64, g1, bb1, st0 + 128, st0 + 192, 32, invN);

  k_conv2<<<8192, 256, 0, stream>>>(yB, st0 + 128, st0 + 192, wpk, yA);
  k_redstats<64><<<2048, 256, 0, stream>>>(yA, st1, st1 + 64);
  k_stats<<<1, 64, 0, stream>>>(st1, st1 + 64, g2, bb2, st1 + 128, st1 + 192, 64, invN);

  k_conv3x3<64, 18><<<8192, 256, 0, stream>>>(yA, st1 + 128, st1 + 192, wpk + 2048, yB);
  k_redstats<32><<<2048, 256, 0, stream>>>(yB, st2, st2 + 64);
  k_stats<<<1, 64, 0, stream>>>(st2, st2 + 64, g3, bb3, st2 + 128, st2 + 192, 32, invN);

  k_conv3x3<32, 9><<<8192, 256, 0, stream>>>(yB, st2 + 128, st2 + 192, wpk + 20480, yA);
  k_redstats<32><<<2048, 256, 0, stream>>>(yA, st3, st3 + 64);
  k_stats<<<1, 64, 0, stream>>>(st3, st3 + 64, g4, bb4, st3 + 128, st3 + 192, 32, invN);

  k_fc<<<8192, 256, 0, stream>>>(yA, st3 + 128, st3 + 192, wpk + 29696, wpk + 30720,
                                 bf1, bf2, y0sum);

  k_head<<<32, 32, 0, stream>>>(y0sum, cmws, giws, out + OFF_WI, out + OFF_VALS,
                                out + OFF_ORDER);
  k_sel<<<122880, 256, 0, stream>>>(x, giws, out);
  k_cgh<<<8192, 256, 0, stream>>>(x, cmws, out + OFF_CGH);
}
